// ReviewClassifierRNN_50405736186257
// MI455X (gfx1250) — matrix operand order repaired and checked by reading, not compiled, not run on hardware
//
#include <hip/hip_runtime.h>
#include <hip/hip_bf16.h>
#include <math.h>

// ---------------------------------------------------------------------------
// ReviewClassifierRNN on gfx1250 (MI455X): 2-layer tanh RNN + FC + sigmoid.
// Persistent-kernel design: the whole 512-step double-layer recurrence runs
// in ONE launch; steps separated by a grid-wide software barrier.
// Each wave owns a fixed 16x16 output tile, so its 32KB weight slice is
// preloaded into 256 VGPRs ONCE and reused for all 512 steps (CDNA5 allows
// 1024 VGPRs/wave). Matmuls use V_WMMA_F32_16X16X32_F16.
// ---------------------------------------------------------------------------

typedef __attribute__((ext_vector_type(16))) _Float16 v16h;
typedef __attribute__((ext_vector_type(8)))  _Float16 v8h;
typedef __attribute__((ext_vector_type(8)))  float    v8f;

#define B_ 64
#define S_ 512
#define F_ 512
#define H_ 1024
#define NBLK 96        // persistent grid: 32 layer-0 blocks + 64 layer-1 blocks
#define KTILES 32      // K=1024 / 32 per WMMA

// ---- WMMA operand loaders (contiguous k order, used by A and B alike) -------------------
// A 16x32 f16 tile: lane<16 -> M=lane, K {0..15};
//                   lane>=16 -> M=lane-16, K {16..31}.
// Caller passes p = &Arow[k0 + hi*16]; second chunk at +8 elements.
__device__ __forceinline__ v16h load_a_f16(const _Float16* p) {
  v8h x0 = *(const v8h*)(p);
  v8h x1 = *(const v8h*)(p + 8);
  v16h a;
#pragma unroll
  for (int i = 0; i < 8; ++i) { a[i] = x0[i]; a[i + 8] = x1[i]; }
  return a;
}

__device__ __forceinline__ v16h load_a_f32(const float* p) {
  v8f x0 = *(const v8f*)(p);
  v8f x1 = *(const v8f*)(p + 8);
  v16h a;
#pragma unroll
  for (int i = 0; i < 8; ++i) { a[i] = (_Float16)x0[i]; a[i + 8] = (_Float16)x1[i]; }
  return a;
}

// ---- f32 -> f16 weight conversion ------------------------------------------
__global__ void cvt_f32_to_f16(const float* __restrict__ src,
                               _Float16* __restrict__ dst, int n) {
  int i = blockIdx.x * blockDim.x + threadIdx.x;
  if (i < n) dst[i] = (_Float16)src[i];
}

// ---- Layer-0 input projection: C[m][n] = x[m][:] . W[n][:] + ba[n] + bb[n] -
__global__ void xproj_wmma(const float* __restrict__ A32, int lda,
                           const _Float16* __restrict__ W, int K,
                           const float* __restrict__ bias_a,
                           const float* __restrict__ bias_b,
                           float* __restrict__ C) {
  const int wave = threadIdx.x >> 5;
  const int lane = threadIdx.x & 31;
  const int hi   = lane >> 4, lo = lane & 15;
  const int m0   = blockIdx.x << 4;
  const int n0   = (blockIdx.y << 7) + (wave << 4);

  v8f acc = {};
  const size_t arow = (size_t)(m0 + lo) * lda;
  const _Float16* wrow = W + (size_t)(n0 + lo) * K + hi * 16;
  for (int k0 = 0; k0 < K; k0 += 32) {
    v16h a = load_a_f32(A32 + arow + k0 + hi * 16);
    v16h b = *(const v16h*)(wrow + k0);   // 32B contiguous: W[n][k0+hi*16 ..]
    acc = __builtin_amdgcn_wmma_f32_16x16x32_f16(
        false, a, false, b, (short)0, acc, false, false);
  }
  const int n = n0 + lo;
  const float bs = bias_a[n] + bias_b[n];
#pragma unroll
  for (int r = 0; r < 8; ++r) {
    const int m = m0 + r + (hi << 3);
    C[(size_t)m * H_ + n] = acc[r] + bs;
  }
}

// ---- Grid-wide barrier (sense via generation counter) ----------------------
__device__ __forceinline__ void grid_barrier(int* bar_cnt, int* bar_gen) {
  __threadfence();        // release: this wave's global stores visible at L2
  __syncthreads();        // whole block done + fenced
  if (threadIdx.x == 0) {
    const int gen = __hip_atomic_load(bar_gen, __ATOMIC_RELAXED,
                                      __HIP_MEMORY_SCOPE_AGENT);
    if (__hip_atomic_fetch_add(bar_cnt, 1, __ATOMIC_ACQ_REL,
                               __HIP_MEMORY_SCOPE_AGENT) == NBLK - 1) {
      __hip_atomic_store(bar_cnt, 0, __ATOMIC_RELAXED,
                         __HIP_MEMORY_SCOPE_AGENT);
      __hip_atomic_fetch_add(bar_gen, 1, __ATOMIC_RELEASE,
                             __HIP_MEMORY_SCOPE_AGENT);
    } else {
      while (__hip_atomic_load(bar_gen, __ATOMIC_ACQUIRE,
                               __HIP_MEMORY_SCOPE_AGENT) == gen)
        __builtin_amdgcn_s_sleep(1);
    }
  }
  __syncthreads();
  __threadfence();        // acquire: drop stale near-cache lines before reads
}

// ---- Persistent pipelined recurrence ---------------------------------------
// Blocks 0..31  (1 tile/wave): layer-0 step t:
//   h1_t = tanh(xp0_t + h1_{t-1} @ Whh0^T)              (256 tiles, K=1024)
// Blocks 32..95 (2 waves/tile): layer-1 step s = t-1:
//   h2_s = tanh(b1 + h1_s @ Wih1^T + h2_{s-1} @ Whh1^T) (LDS pair-reduce)
// Every wave preloads its loop-invariant 32KB weight slice into 256 VGPRs.
__global__ void __launch_bounds__(256)
rnn_persistent(const _Float16* __restrict__ whh0,
               const float*    __restrict__ xp,
               _Float16*       __restrict__ h1,
               const _Float16* __restrict__ hz,
               const _Float16* __restrict__ wih1,
               const _Float16* __restrict__ whh1,
               const float*    __restrict__ bih1,
               const float*    __restrict__ bhh1,
               _Float16*       __restrict__ pg0,
               _Float16*       __restrict__ pg1,
               const float*    __restrict__ fcw,
               const float*    __restrict__ fcb,
               float*          __restrict__ out,
               int* bar_cnt, int* bar_gen) {
  __shared__ float red[8 * 256];
  const int g    = blockIdx.x;
  const int wave = threadIdx.x >> 5;
  const int lane = threadIdx.x & 31;
  const int hi   = lane >> 4, lo = lane & 15;
  const long ldbig = (long)S_ * H_;

  if (g < 32) {
    // ------------------------- layer-0 blocks -------------------------------
    const int q  = (g << 3) + wave;            // 0..255
    const int m0 = (q >> 6) << 4;
    const int n0 = (q & 63) << 4;
    // Preload loop-invariant weight slice: Whh0 rows n0..n0+15, K=0..1023.
    const _Float16* wrow = whh0 + (size_t)(n0 + lo) * H_ + hi * 16;
    v16h breg[KTILES];
#pragma unroll
    for (int i = 0; i < KTILES; ++i) breg[i] = *(const v16h*)(wrow + (i << 5));

    for (int t = 0; t <= S_; ++t) {
      if (t < S_) {
        const _Float16* A   = (t == 0) ? hz : (h1 + (size_t)(t - 1) * H_);
        const long      lda = (t == 0) ? (long)H_ : ldbig;
        const _Float16* arow = A + (size_t)(m0 + lo) * lda + hi * 16;
        v8f acc = {};
#pragma unroll
        for (int i = 0; i < KTILES; ++i) {
          v16h a = load_a_f16(arow + (i << 5));
          acc = __builtin_amdgcn_wmma_f32_16x16x32_f16(
              false, a, false, breg[i], (short)0, acc, false, false);
        }
        const float* xpt = xp + (size_t)t * H_;
        _Float16*    ho  = h1 + (size_t)t * H_;
        const int n = n0 + lo;
#pragma unroll
        for (int r = 0; r < 8; ++r) {
          const int m = m0 + r + (hi << 3);
          float v = tanhf(acc[r] + xpt[(size_t)m * ldbig + n]);
          ho[(size_t)m * ldbig + n] = (_Float16)v;
        }
      }
      grid_barrier(bar_cnt, bar_gen);
    }
  } else {
    // ------------------------- layer-1 blocks -------------------------------
    const int q    = ((g - 32) << 2) + (wave >> 1);   // 0..255
    const int half = wave & 1;                        // 0: Wih1, 1: Whh1
    const int m0   = (q >> 6) << 4;
    const int n0   = (q & 63) << 4;
    const _Float16* W = half ? whh1 : wih1;
    const _Float16* wrow = W + (size_t)(n0 + lo) * H_ + hi * 16;
    v16h breg[KTILES];
#pragma unroll
    for (int i = 0; i < KTILES; ++i) breg[i] = *(const v16h*)(wrow + (i << 5));

    for (int t = 0; t <= S_; ++t) {
      if (t >= 1) {
        const int s = t - 1;
        const _Float16* h2prev = (s & 1) ? pg1 : pg0;   // s=0 reads zeroed pg0
        _Float16*       h2out  = (s & 1) ? pg0 : pg1;
        const _Float16* A   = half ? h2prev : (h1 + (size_t)s * H_);
        const long      lda = half ? (long)H_ : ldbig;
        const _Float16* arow = A + (size_t)(m0 + lo) * lda + hi * 16;
        v8f acc = {};
#pragma unroll
        for (int i = 0; i < KTILES; ++i) {
          v16h a = load_a_f16(arow + (i << 5));
          acc = __builtin_amdgcn_wmma_f32_16x16x32_f16(
              false, a, false, breg[i], (short)0, acc, false, false);
        }
        float* myred = red + (wave << 8);
#pragma unroll
        for (int r = 0; r < 8; ++r) myred[(r << 5) + lane] = acc[r];
        __syncthreads();
        if (!half) {
          const float* rw0 = red + (wave << 8);
          const float* rw1 = rw0 + 256;
          const int n = n0 + lo;
          const float bs = bih1[n] + bhh1[n];
#pragma unroll
          for (int r = 0; r < 8; ++r) {
            const int m = m0 + r + (hi << 3);
            float v = tanhf(rw0[(r << 5) + lane] + rw1[(r << 5) + lane] + bs);
            h2out[(size_t)m * H_ + n] = (_Float16)v;
          }
        }
      }
      grid_barrier(bar_cnt, bar_gen);
    }
  }

  // Final FC (H -> 1) + sigmoid on block 0. s=511 (odd) wrote pg0.
  if (g == 0) {
    const int b = threadIdx.x;
    if (b < B_) {
      const _Float16* h = pg0 + (size_t)b * H_;
      float ssum = 0.f;
      for (int k = 0; k < H_; ++k) ssum += (float)h[k] * fcw[k];
      ssum += fcb[0];
      out[b] = 1.0f / (1.0f + expf(-ssum));
    }
  }
}

// ---------------------------------------------------------------------------
extern "C" void kernel_launch(void* const* d_in, const int* in_sizes, int n_in,
                              void* d_out, int out_size, void* d_ws, size_t ws_size,
                              hipStream_t stream) {
  (void)in_sizes; (void)n_in; (void)out_size; (void)ws_size;
  const float* x     = (const float*)d_in[0];
  const float* W_ih0 = (const float*)d_in[1];
  const float* W_hh0 = (const float*)d_in[2];
  const float* b_ih0 = (const float*)d_in[3];
  const float* b_hh0 = (const float*)d_in[4];
  const float* W_ih1 = (const float*)d_in[5];
  const float* W_hh1 = (const float*)d_in[6];
  const float* b_ih1 = (const float*)d_in[7];
  const float* b_hh1 = (const float*)d_in[8];
  const float* fc_w  = (const float*)d_in[9];
  const float* fc_b  = (const float*)d_in[10];
  float* out = (float*)d_out;

  // Workspace carve-up (256B aligned).
  char*  ws  = (char*)d_ws;
  size_t off = 0;
  auto carve = [&](size_t bytes) -> char* {
    char* p = ws + off;
    off = (off + bytes + 255) & ~(size_t)255;
    return p;
  };
  _Float16* wih0 = (_Float16*)carve((size_t)H_ * F_ * 2);
  _Float16* whh0 = (_Float16*)carve((size_t)H_ * H_ * 2);
  _Float16* wih1 = (_Float16*)carve((size_t)H_ * H_ * 2);
  _Float16* whh1 = (_Float16*)carve((size_t)H_ * H_ * 2);
  float*    xp   = (float*)   carve((size_t)B_ * S_ * H_ * 4);  // layer-0 xproj
  _Float16* h1   = (_Float16*)carve((size_t)B_ * S_ * H_ * 2);  // layer-0 outputs
  _Float16* hz   = (_Float16*)carve((size_t)B_ * H_ * 2);       // zero h0
  _Float16* pg0  = (_Float16*)carve((size_t)B_ * H_ * 2);       // layer-1 ping
  _Float16* pg1  = (_Float16*)carve((size_t)B_ * H_ * 2);       // layer-1 pong
  int*      bar  = (int*)     carve(256);                       // [cnt, gen]

  // Weight conversion (deterministic, every launch) + zero-init states.
  cvt_f32_to_f16<<<(H_ * F_ + 255) / 256, 256, 0, stream>>>(W_ih0, wih0, H_ * F_);
  cvt_f32_to_f16<<<(H_ * H_ + 255) / 256, 256, 0, stream>>>(W_hh0, whh0, H_ * H_);
  cvt_f32_to_f16<<<(H_ * H_ + 255) / 256, 256, 0, stream>>>(W_ih1, wih1, H_ * H_);
  cvt_f32_to_f16<<<(H_ * H_ + 255) / 256, 256, 0, stream>>>(W_hh1, whh1, H_ * H_);
  hipMemsetAsync(hz,  0, (size_t)B_ * H_ * 2, stream);
  hipMemsetAsync(pg0, 0, (size_t)B_ * H_ * 2, stream);
  hipMemsetAsync(bar, 0, 256, stream);

  // Layer-0 input projection (fully parallel big GEMM).
  xproj_wmma<<<dim3(B_ * S_ / 16, H_ / 128), 256, 0, stream>>>(
      x, F_, wih0, F_, b_ih0, b_hh0, xp);

  // Entire double-layer recurrence + FC in one persistent launch.
  rnn_persistent<<<NBLK, 256, 0, stream>>>(
      whh0, xp, h1, hz, wih1, whh1, b_ih1, b_hh1,
      pg0, pg1, fc_w, fc_b, out, bar, bar + 1);
}
